// GRU_44719199486504
// MI455X (gfx1250) — matrix operand order repaired and checked by reading, not compiled, not run on hardware
//
#include <hip/hip_runtime.h>
#include <hip/hip_bf16.h>
#include <cstdint>

// ---------------------------------------------------------------------------
// GRU on gfx1250: bf16 WMMA GEMMs + fused gate epilogues.
//   x:  [B=128, T=256, D=512]  f32
//   Wz/Wr/Wi: [L=1024, D+L=1536] f32, bz/br/bi: [L] f32
//   out: h_hist [T, B, L] f32
// ---------------------------------------------------------------------------

typedef __bf16 bf16_t;
typedef __attribute__((ext_vector_type(16))) __bf16 v16bf;
typedef __attribute__((ext_vector_type(8)))  __bf16 v8bf;
typedef __attribute__((ext_vector_type(8)))  float  v8f;

#define BATCH 128
#define SEQ   256
#define DIM   512
#define LAT   1024
#define KTOT  (DIM + LAT)   // 1536

// Build a 16-element bf16 fragment from two contiguous 16-byte chunks.
__device__ __forceinline__ v16bf load_frag(const bf16_t* p0, const bf16_t* p1) {
  v8bf lo = *reinterpret_cast<const v8bf*>(p0);
  v8bf hi = *reinterpret_cast<const v8bf*>(p1);
  v16bf r;
#pragma unroll
  for (int i = 0; i < 8; ++i) { r[i] = lo[i]; r[i + 8] = hi[i]; }
  return r;
}

__device__ __forceinline__ float sigmoid_f(float x) {
  return 1.0f / (1.0f + __expf(-x));
}

// ---------------------------------------------------------------------------
// Phase A: z = sigmoid([x_t,h] Wz^T + bz) ; rh = sigmoid([x_t,h] Wr^T + br)*h
// grid = 128 blocks (gate = blk>>6; 8 m-tiles x 8 n-chunks), 8 waves/block,
// one 16x16 C tile per wave, K-loop 1536 in steps of 32 (bf16 WMMA).
// ---------------------------------------------------------------------------
__global__ __launch_bounds__(256) void gru_phaseA(
    const bf16_t* __restrict__ xbf,   // [B, T, D] bf16
    const bf16_t* __restrict__ hbf,   // [B, L]    bf16 (h_{t-1})
    const float*  __restrict__ hf32,  // [B, L]    f32  (h_{t-1})
    const bf16_t* __restrict__ wz,    // [L, KTOT] bf16
    const bf16_t* __restrict__ wr,    // [L, KTOT] bf16
    const float*  __restrict__ bz,
    const float*  __restrict__ br,
    float*        __restrict__ zf32,  // [B, L] f32
    bf16_t*       __restrict__ rhbf,  // [B, L] bf16 (r * h)
    int t)
{
  const int lane  = threadIdx.x & 31;
  const int wave  = threadIdx.x >> 5;
  const int gate  = blockIdx.x >> 6;            // 0 = z, 1 = r
  const int rem   = blockIdx.x & 63;
  const int m0    = (rem >> 3) * 16;            // batch tile base
  const int n0    = ((rem & 7) * 8 + wave) * 16;// latent tile base

  const bf16_t* W    = gate ? wr : wz;
  const float*  bias = gate ? br : bz;

  const int lrow  = lane & 15;
  const int khalf = lane >> 4;
  const int koffA = khalf * 8;    // A: lanes 0-15 K 0..7/16..23; 16-31 K 8..15/24..31
  const int koffB = khalf * 8;    // B: lanes 0-15 K 0..7/16..23; 16-31 K 8..15/24..31

  const int bb = m0 + lrow;       // batch row of A fragment
  const int nn = n0 + lrow;       // output column of B fragment

  const bf16_t* xrow = xbf + ((size_t)bb * SEQ + t) * DIM;
  const bf16_t* hrow = hbf + (size_t)bb * LAT;
  const bf16_t* wrow = W   + (size_t)nn * KTOT;

  v8f c = {};

  // K in [0, DIM): A comes from x_t
#pragma unroll 4
  for (int k0 = 0; k0 < DIM; k0 += 32) {
    if (k0 + 256 < KTOT) __builtin_prefetch(wrow + k0 + koffB + 256, 0, 1);
    v16bf a = load_frag(xrow + k0 + koffA, xrow + k0 + 16 + koffA);
    const bf16_t* wp = wrow + k0 + koffB;
    v16bf b = load_frag(wp, wp + 16);
    c = __builtin_amdgcn_wmma_f32_16x16x32_bf16(false, a, false, b,
                                                (short)0, c, false, false);
  }
  // K in [DIM, KTOT): A comes from h_{t-1}
#pragma unroll 4
  for (int k0 = 0; k0 < LAT; k0 += 32) {
    v16bf a = load_frag(hrow + k0 + koffA, hrow + k0 + 16 + koffA);
    const bf16_t* wp = wrow + DIM + k0 + koffB;
    v16bf b = load_frag(wp, wp + 16);
    c = __builtin_amdgcn_wmma_f32_16x16x32_bf16(false, a, false, b,
                                                (short)0, c, false, false);
  }

  const float bv = bias[nn];
#pragma unroll
  for (int r = 0; r < 8; ++r) {
    const int    mb  = m0 + r + khalf * 8;      // C/D: VGPR r -> M=r (lanes<16) / r+8
    const size_t idx = (size_t)mb * LAT + nn;
    const float  g   = sigmoid_f(c[r] + bv);
    if (gate == 0) zf32[idx] = g;
    else           rhbf[idx] = (bf16_t)(g * hf32[idx]);
  }
}

// ---------------------------------------------------------------------------
// Phase B: h' = (1-z)*h + z*tanh([x_t, rh] Wi^T + bi); write h_hist[t] (f32)
// and refresh h (f32 + bf16) for the next step. grid = 64 blocks, 8 waves.
// ---------------------------------------------------------------------------
__global__ __launch_bounds__(256) void gru_phaseB(
    const bf16_t* __restrict__ xbf,
    const bf16_t* __restrict__ rhbf,  // [B, L] bf16
    const bf16_t* __restrict__ wi,    // [L, KTOT] bf16
    const float*  __restrict__ bi,
    const float*  __restrict__ zf32,  // [B, L]
    float*        __restrict__ hf32,  // [B, L] in: h_{t-1}, out: h_t
    bf16_t*       __restrict__ hbf,   // [B, L] out: h_t
    float*        __restrict__ out,   // [T, B, L]
    int t)
{
  const int lane  = threadIdx.x & 31;
  const int wave  = threadIdx.x >> 5;
  const int m0    = (blockIdx.x >> 3) * 16;
  const int n0    = ((blockIdx.x & 7) * 8 + wave) * 16;

  const int lrow  = lane & 15;
  const int khalf = lane >> 4;
  const int koffA = khalf * 8;
  const int koffB = khalf * 8;

  const int bb = m0 + lrow;
  const int nn = n0 + lrow;

  const bf16_t* xrow  = xbf  + ((size_t)bb * SEQ + t) * DIM;
  const bf16_t* rhrow = rhbf + (size_t)bb * LAT;
  const bf16_t* wrow  = wi   + (size_t)nn * KTOT;

  v8f c = {};

#pragma unroll 4
  for (int k0 = 0; k0 < DIM; k0 += 32) {
    if (k0 + 256 < KTOT) __builtin_prefetch(wrow + k0 + koffB + 256, 0, 1);
    v16bf a = load_frag(xrow + k0 + koffA, xrow + k0 + 16 + koffA);
    const bf16_t* wp = wrow + k0 + koffB;
    v16bf b = load_frag(wp, wp + 16);
    c = __builtin_amdgcn_wmma_f32_16x16x32_bf16(false, a, false, b,
                                                (short)0, c, false, false);
  }
#pragma unroll 4
  for (int k0 = 0; k0 < LAT; k0 += 32) {
    v16bf a = load_frag(rhrow + k0 + koffA, rhrow + k0 + 16 + koffA);
    const bf16_t* wp = wrow + DIM + k0 + koffB;
    v16bf b = load_frag(wp, wp + 16);
    c = __builtin_amdgcn_wmma_f32_16x16x32_bf16(false, a, false, b,
                                                (short)0, c, false, false);
  }

  const float bv = bi[nn];
#pragma unroll
  for (int r = 0; r < 8; ++r) {
    const int    mb  = m0 + r + khalf * 8;
    const size_t idx = (size_t)mb * LAT + nn;
    const float  cand = tanhf(c[r] + bv);
    const float  z    = zf32[idx];
    const float  hp   = hf32[idx];
    const float  hn   = (1.0f - z) * hp + z * cand;
    out[((size_t)t * BATCH + mb) * LAT + nn] = hn;
    hf32[idx] = hn;
    hbf[idx]  = (bf16_t)hn;
  }
}

// ---------------------------------------------------------------------------
// Prologue kernels
// ---------------------------------------------------------------------------
__global__ void f32_to_bf16_kernel(const float* __restrict__ in,
                                   bf16_t* __restrict__ out, int n) {
  for (int i = blockIdx.x * blockDim.x + threadIdx.x; i < n;
       i += gridDim.x * blockDim.x)
    out[i] = (bf16_t)in[i];
}

__global__ void zero_h_kernel(float* __restrict__ hf32,
                              bf16_t* __restrict__ hbf) {
  const int i = blockIdx.x * blockDim.x + threadIdx.x;
  if (i < BATCH * LAT) { hf32[i] = 0.0f; hbf[i] = (bf16_t)0.0f; }
}

// ---------------------------------------------------------------------------
extern "C" void kernel_launch(void* const* d_in, const int* in_sizes, int n_in,
                              void* d_out, int out_size, void* d_ws, size_t ws_size,
                              hipStream_t stream) {
  (void)in_sizes; (void)n_in; (void)out_size; (void)ws_size;

  const float* x  = (const float*)d_in[0];
  const float* Wz = (const float*)d_in[1];
  const float* bz = (const float*)d_in[2];
  const float* Wr = (const float*)d_in[3];
  const float* br = (const float*)d_in[4];
  const float* Wi = (const float*)d_in[5];
  const float* bi = (const float*)d_in[6];
  float* out = (float*)d_out;

  // ---- workspace carve-out (256B aligned), ~42.5 MB total ----
  char*  ws  = (char*)d_ws;
  size_t off = 0;
  auto carve = [&](size_t bytes) -> char* {
    char* p = ws + off;
    off = (off + bytes + 255) & ~(size_t)255;
    return p;
  };
  bf16_t* wzbf = (bf16_t*)carve((size_t)LAT * KTOT * sizeof(bf16_t));
  bf16_t* wrbf = (bf16_t*)carve((size_t)LAT * KTOT * sizeof(bf16_t));
  bf16_t* wibf = (bf16_t*)carve((size_t)LAT * KTOT * sizeof(bf16_t));
  bf16_t* xbf  = (bf16_t*)carve((size_t)BATCH * SEQ * DIM * sizeof(bf16_t));
  float*  hf32 = (float*) carve((size_t)BATCH * LAT * sizeof(float));
  bf16_t* hbf  = (bf16_t*)carve((size_t)BATCH * LAT * sizeof(bf16_t));
  float*  zf32 = (float*) carve((size_t)BATCH * LAT * sizeof(float));
  bf16_t* rhbf = (bf16_t*)carve((size_t)BATCH * LAT * sizeof(bf16_t));

  // ---- one-time conversions + h0 = 0 ----
  const int wn = LAT * KTOT;                 // 1,572,864
  const int xn = BATCH * SEQ * DIM;          // 16,777,216
  f32_to_bf16_kernel<<<2048, 256, 0, stream>>>(Wz, wzbf, wn);
  f32_to_bf16_kernel<<<2048, 256, 0, stream>>>(Wr, wrbf, wn);
  f32_to_bf16_kernel<<<2048, 256, 0, stream>>>(Wi, wibf, wn);
  f32_to_bf16_kernel<<<4096, 256, 0, stream>>>(x,  xbf,  xn);
  zero_h_kernel<<<(BATCH * LAT + 255) / 256, 256, 0, stream>>>(hf32, hbf);

  // ---- sequential recurrence: 2 kernels per timestep ----
  for (int t = 0; t < SEQ; ++t) {
    gru_phaseA<<<128, 256, 0, stream>>>(xbf, hbf, hf32, wzbf, wrbf,
                                        bz, br, zf32, rhbf, t);
    gru_phaseB<<<64, 256, 0, stream>>>(xbf, rhbf, wibf, bi, zf32,
                                       hf32, hbf, out, t);
  }
}
